// DingoT1Model_35656818491575
// MI455X (gfx1250) — compile-verified, matrix operand order checked by reading, not run on hardware
//
#include <hip/hip_runtime.h>
#include <hip/hip_bf16.h>
#include <math.h>

// ---------------------------------------------------------------------------
// Model constants (reference): B=64, L=256, seq=257 (pad 288), d=128, dff=256,
// 2 heads x dh=64, ctx=32, n_params=11, S=4096.
// ---------------------------------------------------------------------------
#define BATCH   64
#define LTOK    256
#define SEQ     257
#define SEQP    288      // 18 tiles of 16; rows >=257 zero-padded
#define DMODEL  128
#define DFF     256
#define DH      64
#define CTXD    32
#define NP      11
#define NS      4096
#define NROWS   (BATCH * NS)

typedef _Float16 half_t;
typedef _Float16 v16h __attribute__((ext_vector_type(16)));
typedef _Float16 v8h  __attribute__((ext_vector_type(8)));
typedef float    v8f  __attribute__((ext_vector_type(8)));

union FragH { v16h v; v8h h[2]; half_t e[16]; };

// Wave-local LDS fence: CDNA5 split DS counter + compiler barrier.
__device__ __forceinline__ void wfence() {
  asm volatile("s_wait_dscnt 0" ::: "memory");
}

// 16-bit A-matrix 16x32 K index map (ISA 7.12.2): VGPR h/2, lane group g.
__device__ __forceinline__ int kmap(int h, int g) {
  return (h < 8) ? (g * 8 + h) : (16 + g * 8 + (h - 8));
}

// Vectorized A-style fragment from row-major f16 (also used for W (N,K)):
// per lane two contiguous 16-byte runs -> two b128 loads. Requires ld % 8 == 0
// and 16B-aligned base (all our buffers satisfy this).
__device__ __forceinline__ v16h fragA_f16(const half_t* src, int ld, int lane) {
  const int m = lane & 15, g = lane >> 4;
  const half_t* p = src + (size_t)m * ld + g * 8;
  FragH f;
  f.h[0] = *(const v8h*)(p);        // K = g*8 + 0..7
  f.h[1] = *(const v8h*)(p + 16);   // K = 16 + g*8 + 0..7
  return f.v;
}

__device__ __forceinline__ v8f wmma(v16h a, v16h b, v8f c) {
  return __builtin_amdgcn_wmma_f32_16x16x32_f16(false, a, false, b, (short)0, c,
                                                false, false);
}

// ---------------------------------------------------------------------------
// K0a: generic f32 -> f16 conversion (weights / tokens), grid-stride.
// ---------------------------------------------------------------------------
__global__ void k_cvt(const float* __restrict__ src, half_t* __restrict__ dst,
                      int n) {
  int i = blockIdx.x * blockDim.x + threadIdx.x;
  const int stride = gridDim.x * blockDim.x;
  for (; i < n; i += stride) dst[i] = (half_t)src[i];
}

// ---------------------------------------------------------------------------
// K0b: pack flow weights to padded f16 (K padded to 64, W3 N padded to 16).
// ---------------------------------------------------------------------------
__global__ void k_pack(const float* W1a, const float* W2a, const float* W3a,
                       const float* W1b, const float* W2b, const float* W3b,
                       half_t* W1pa, half_t* W2pa, half_t* W3pa,
                       half_t* W1pb, half_t* W2pb, half_t* W3pb) {
  const int t = blockIdx.x * blockDim.x + threadIdx.x;
  const int stride = gridDim.x * blockDim.x;
  for (int i = t; i < 128 * 64; i += stride) {
    int r = i >> 6, c = i & 63;
    W1pa[i] = (half_t)((c < 37) ? W1a[r * 37 + c] : 0.f);
    W1pb[i] = (half_t)((c < 38) ? W1b[r * 38 + c] : 0.f);
  }
  for (int i = t; i < 128 * 128; i += stride) {
    W2pa[i] = (half_t)W2a[i];
    W2pb[i] = (half_t)W2b[i];
  }
  for (int i = t; i < 16 * 128; i += stride) {
    int r = i >> 7;
    W3pa[i] = (half_t)((r < 12) ? W3a[i] : 0.f);
    W3pb[i] = (half_t)((r < 10) ? W3b[i] : 0.f);
  }
}

// ---------------------------------------------------------------------------
// K1: embedding. grid(16, B), block 128 (4 waves). 16 token rows per block.
// e = relu(tok @ W1^T + b1) @ W2^T + b2 -> x/xh[b][1+row][:]; block(0,b)
// also writes the summary row.
// ---------------------------------------------------------------------------
__global__ void k_embed(const half_t* __restrict__ tok16,
                        const half_t* __restrict__ W1p, const float* __restrict__ b1,
                        const half_t* __restrict__ W2p, const float* __restrict__ b2,
                        const float* __restrict__ summary,
                        float* __restrict__ x, half_t* __restrict__ xh) {
  extern __shared__ char smem[];
  half_t* h1s = (half_t*)smem;  // [16][256] f16
  const int b = blockIdx.y, t0 = blockIdx.x * 16;
  const int wave = threadIdx.x >> 5, lane = threadIdx.x & 31;
  const int nn = lane & 15, g = lane >> 4;
  const half_t* tok = tok16 + ((size_t)b * LTOK + t0) * 96;

  for (int ni = 0; ni < 4; ++ni) {            // h1: 16 N-tiles of 256
    int n = wave * 4 + ni;
    v8f acc = {};
    for (int kk = 0; kk < 3; ++kk) {          // K = 96
      v16h a = fragA_f16(tok + kk * 32, 96, lane);
      v16h w = fragA_f16(W1p + (size_t)(n * 16) * 96 + kk * 32, 96, lane);
      acc = wmma(a, w, acc);
    }
    float bias = b1[n * 16 + nn];
#pragma unroll
    for (int r = 0; r < 8; ++r) {
      float v = acc[r] + bias;
      h1s[(r + 8 * g) * DFF + n * 16 + nn] = (half_t)(v > 0.f ? v : 0.f);
    }
  }
  __syncthreads();
  for (int ni = 0; ni < 2; ++ni) {            // e: 8 N-tiles of 128
    int n = wave * 2 + ni;
    v8f acc = {};
    for (int kk = 0; kk < 8; ++kk) {          // K = 256
      v16h a = fragA_f16(h1s + kk * 32, DFF, lane);
      v16h w = fragA_f16(W2p + (size_t)(n * 16) * DFF + kk * 32, DFF, lane);
      acc = wmma(a, w, acc);
    }
    float bias = b2[n * 16 + nn];
    size_t base = ((size_t)b * SEQP + 1 + t0) * DMODEL + n * 16 + nn;
#pragma unroll
    for (int r = 0; r < 8; ++r) {
      float vv = acc[r] + bias;
      x[base + (size_t)(r + 8 * g) * DMODEL] = vv;
      xh[base + (size_t)(r + 8 * g) * DMODEL] = (half_t)vv;
    }
  }
  if (blockIdx.x == 0 && threadIdx.x < DMODEL) {
    float sv = summary[threadIdx.x];
    x[(size_t)b * SEQP * DMODEL + threadIdx.x] = sv;
    xh[(size_t)b * SEQP * DMODEL + threadIdx.x] = (half_t)sv;
  }
}

// ---------------------------------------------------------------------------
// K2: qkv = xh @ Wqkv^T + b -> f16 q/k row-major, v transposed (head-major,
// [b][head*64+d][seq]). grid(17, B), block 128.
// ---------------------------------------------------------------------------
__global__ void k_qkv(const half_t* __restrict__ xh, const half_t* __restrict__ Wqkvp,
                      const float* __restrict__ bqkv,
                      half_t* __restrict__ q, half_t* __restrict__ k,
                      half_t* __restrict__ vT) {
  const int b = blockIdx.y, r0 = blockIdx.x * 16;
  const int wave = threadIdx.x >> 5, lane = threadIdx.x & 31;
  const int nn = lane & 15, g = lane >> 4;
  const half_t* xr = xh + ((size_t)b * SEQP + r0) * DMODEL;
  for (int ni = 0; ni < 6; ++ni) {            // 24 N-tiles of 384
    int n = wave * 6 + ni;
    v8f acc = {};
    for (int kk = 0; kk < 4; ++kk) {          // K = 128
      v16h a = fragA_f16(xr + kk * 32, DMODEL, lane);
      v16h w = fragA_f16(Wqkvp + (size_t)(n * 16) * DMODEL + kk * 32, DMODEL, lane);
      acc = wmma(a, w, acc);
    }
    float bias = bqkv[n * 16 + nn];
    int col = (n & 7) * 16 + nn;
    if (n < 16) {                             // q, k: row-major
      half_t* dstbuf = (n < 8) ? q : k;
#pragma unroll
      for (int r = 0; r < 8; ++r) {
        int row = r0 + r + 8 * g;
        if (row < SEQ)
          dstbuf[((size_t)b * SEQP + row) * DMODEL + col] = (half_t)(acc[r] + bias);
      }
    } else {                                  // v: transposed per head
#pragma unroll
      for (int r = 0; r < 8; ++r) {
        int row = r0 + r + 8 * g;
        if (row < SEQ)
          vT[((size_t)b * DMODEL + col) * SEQP + row] = (half_t)(acc[r] + bias);
      }
    }
  }
}

// ---------------------------------------------------------------------------
// K3: attention, one wave per (b, head, 16-query tile). Scores + softmax in
// LDS, then P @ V (V pre-transposed). Keys rows >= 257 are zero and masked.
// ---------------------------------------------------------------------------
__global__ void k_attn(const half_t* __restrict__ q, const half_t* __restrict__ k,
                       const half_t* __restrict__ vT,
                       const unsigned char* __restrict__ mask,
                       half_t* __restrict__ o) {
  extern __shared__ char smem[];
  float*  sc = (float*)smem;                      // [16][SEQP]
  half_t* pb = (half_t*)(smem + 16 * SEQP * 4);   // [16][SEQP]
  const int b = blockIdx.z, h = blockIdx.y, q0 = blockIdx.x * 16;
  const int lane = threadIdx.x & 31;
  const int nn = lane & 15, g = lane >> 4;
  const half_t* qh = q + ((size_t)b * SEQP + q0) * DMODEL + h * DH;

  for (int kt = 0; kt < SEQP / 16; ++kt) {        // scores: 18 key tiles
    v8f acc = {};
#pragma unroll
    for (int kk = 0; kk < 2; ++kk) {              // K = dh = 64
      v16h a  = fragA_f16(qh + kk * 32, DMODEL, lane);
      v16h bb = fragA_f16(k + ((size_t)b * SEQP + kt * 16) * DMODEL + h * DH + kk * 32,
                          DMODEL, lane);
      acc = wmma(a, bb, acc);
    }
    int col = kt * 16 + nn;
    bool valid = (col == 0) || (col < SEQ && mask[b * LTOK + (col - 1)] != 0);
#pragma unroll
    for (int r = 0; r < 8; ++r)
      sc[(r + 8 * g) * SEQP + col] = valid ? acc[r] * 0.125f : -1.0e9f;
  }
  wfence();
  if (lane < 16) {                                // softmax, one row per lane
    float* row = sc + lane * SEQP;
    float m = -3.4e38f;
    for (int j = 0; j < SEQP; ++j) m = fmaxf(m, row[j]);
    float s = 0.f;
    for (int j = 0; j < SEQP; ++j) s += __expf(row[j] - m);
    float inv = 1.f / s;
    half_t* prow = pb + lane * SEQP;
    for (int j = 0; j < SEQP; ++j) prow[j] = (half_t)(__expf(row[j] - m) * inv);
  }
  wfence();
  for (int n = 0; n < 4; ++n) {                   // o = P @ V: 4 N-tiles of 64
    v8f acc = {};
    for (int kk = 0; kk < SEQP / 32; ++kk) {      // K = 288
      v16h a  = fragA_f16(pb + kk * 32, SEQP, lane);
      v16h bb = fragA_f16(vT + ((size_t)b * DMODEL + h * DH + n * 16) * SEQP + kk * 32,
                          SEQP, lane);
      acc = wmma(a, bb, acc);
    }
#pragma unroll
    for (int r = 0; r < 8; ++r) {
      int row = q0 + r + 8 * g;
      if (row < SEQ)
        o[((size_t)b * SEQP + row) * DMODEL + h * DH + n * 16 + nn] = (half_t)acc[r];
    }
  }
}

// ---------------------------------------------------------------------------
// K4: x = LN(x + o @ Wo^T + bo), also refresh f16 mirror. grid(17, B).
// ---------------------------------------------------------------------------
__global__ void k_projln(const half_t* __restrict__ o, const half_t* __restrict__ Wop,
                         const float* __restrict__ bo, const float* __restrict__ lng,
                         const float* __restrict__ lnb,
                         float* __restrict__ x, half_t* __restrict__ xh) {
  extern __shared__ char smem[];
  float* sbuf = (float*)smem;  // [16][128]
  const int b = blockIdx.y, r0 = blockIdx.x * 16;
  const int wave = threadIdx.x >> 5, lane = threadIdx.x & 31;
  const int nn = lane & 15, g = lane >> 4;
  const half_t* orow = o + ((size_t)b * SEQP + r0) * DMODEL;
  for (int ni = 0; ni < 2; ++ni) {
    int n = wave * 2 + ni;
    v8f acc = {};
    for (int kk = 0; kk < 4; ++kk) {
      v16h a = fragA_f16(orow + kk * 32, DMODEL, lane);
      v16h w = fragA_f16(Wop + (size_t)(n * 16) * DMODEL + kk * 32, DMODEL, lane);
      acc = wmma(a, w, acc);
    }
    float bias = bo[n * 16 + nn];
#pragma unroll
    for (int r = 0; r < 8; ++r) sbuf[(r + 8 * g) * DMODEL + n * 16 + nn] = acc[r] + bias;
  }
  __syncthreads();
  if (threadIdx.x < 16) {
    int row = r0 + threadIdx.x;
    if (row < SEQ) {
      float* xr = x + ((size_t)b * SEQP + row) * DMODEL;
      half_t* xhr = xh + ((size_t)b * SEQP + row) * DMODEL;
      float* sr = sbuf + threadIdx.x * DMODEL;
      float mu = 0.f;
      for (int j = 0; j < DMODEL; ++j) { sr[j] += xr[j]; mu += sr[j]; }
      mu *= (1.0f / DMODEL);
      float var = 0.f;
      for (int j = 0; j < DMODEL; ++j) { float d = sr[j] - mu; var += d * d; }
      var *= (1.0f / DMODEL);
      float inv = rsqrtf(var + 1e-5f);
      for (int j = 0; j < DMODEL; ++j) {
        float vv = (sr[j] - mu) * inv * lng[j] + lnb[j];
        xr[j] = vv;
        xhr[j] = (half_t)vv;
      }
    }
  }
}

// ---------------------------------------------------------------------------
// K5: x = LN(x + relu(xh @ Wf1^T + bf1) @ Wf2^T + bf2). grid(17, B).
// ---------------------------------------------------------------------------
__global__ void k_ffn(const half_t* __restrict__ Wf1p, const float* __restrict__ bf1,
                      const half_t* __restrict__ Wf2p, const float* __restrict__ bf2,
                      const float* __restrict__ lng, const float* __restrict__ lnb,
                      float* __restrict__ x, half_t* __restrict__ xh) {
  extern __shared__ char smem[];
  half_t* h1s  = (half_t*)smem;                   // [16][256]
  float*  sbuf = (float*)(smem + 16 * DFF * 2);   // [16][128]
  const int b = blockIdx.y, r0 = blockIdx.x * 16;
  const int wave = threadIdx.x >> 5, lane = threadIdx.x & 31;
  const int nn = lane & 15, g = lane >> 4;
  const half_t* xr = xh + ((size_t)b * SEQP + r0) * DMODEL;
  for (int ni = 0; ni < 4; ++ni) {
    int n = wave * 4 + ni;
    v8f acc = {};
    for (int kk = 0; kk < 4; ++kk) {
      v16h a = fragA_f16(xr + kk * 32, DMODEL, lane);
      v16h w = fragA_f16(Wf1p + (size_t)(n * 16) * DMODEL + kk * 32, DMODEL, lane);
      acc = wmma(a, w, acc);
    }
    float bias = bf1[n * 16 + nn];
#pragma unroll
    for (int r = 0; r < 8; ++r) {
      float vv = acc[r] + bias;
      h1s[(r + 8 * g) * DFF + n * 16 + nn] = (half_t)(vv > 0.f ? vv : 0.f);
    }
  }
  __syncthreads();
  for (int ni = 0; ni < 2; ++ni) {
    int n = wave * 2 + ni;
    v8f acc = {};
    for (int kk = 0; kk < 8; ++kk) {
      v16h a = fragA_f16(h1s + kk * 32, DFF, lane);
      v16h w = fragA_f16(Wf2p + (size_t)(n * 16) * DFF + kk * 32, DFF, lane);
      acc = wmma(a, w, acc);
    }
    float bias = bf2[n * 16 + nn];
#pragma unroll
    for (int r = 0; r < 8; ++r) sbuf[(r + 8 * g) * DMODEL + n * 16 + nn] = acc[r] + bias;
  }
  __syncthreads();
  if (threadIdx.x < 16) {
    int row = r0 + threadIdx.x;
    if (row < SEQ) {
      float* xw = x + ((size_t)b * SEQP + row) * DMODEL;
      half_t* xhw = xh + ((size_t)b * SEQP + row) * DMODEL;
      float* sr = sbuf + threadIdx.x * DMODEL;
      float mu = 0.f;
      for (int j = 0; j < DMODEL; ++j) { sr[j] += xw[j]; mu += sr[j]; }
      mu *= (1.0f / DMODEL);
      float var = 0.f;
      for (int j = 0; j < DMODEL; ++j) { float d = sr[j] - mu; var += d * d; }
      var *= (1.0f / DMODEL);
      float inv = rsqrtf(var + 1e-5f);
      for (int j = 0; j < DMODEL; ++j) {
        float vv = (sr[j] - mu) * inv * lng[j] + lnb[j];
        xw[j] = vv;
        xhw[j] = (half_t)vv;
      }
    }
  }
}

// ---------------------------------------------------------------------------
// K6: ctx[b] = x[b,0,:] @ ctxW^T + ctxb. grid 64, block 32.
// ---------------------------------------------------------------------------
__global__ void k_ctx(const float* __restrict__ x, const float* __restrict__ W,
                      const float* __restrict__ bias, float* __restrict__ ctx) {
  const int b = blockIdx.x, o = threadIdx.x;
  const float* xr = x + (size_t)b * SEQP * DMODEL;
  float s = bias[o];
  for (int j = 0; j < DMODEL; ++j) s += xr[j] * W[o * DMODEL + j];
  ctx[b * CTXD + o] = s;
}

// ---------------------------------------------------------------------------
// K7: coupling flow. One wave per 16 rows; two stages of
// relu(A@W1)->relu(@W2)->@W3, coupling update in per-wave LDS tiles.
// grid 4096, block 128 (4 waves), 40 KB dynamic LDS.
// ---------------------------------------------------------------------------
__global__ void k_flow(const float* __restrict__ z, const float* __restrict__ ctx,
                       const half_t* W1pa, const half_t* W2pa, const half_t* W3pa,
                       const float* b1a, const float* b2a, const float* b3a,
                       const half_t* W1pb, const half_t* W2pb, const half_t* W3pb,
                       const float* b1b, const float* b2b, const float* b3b,
                       float* __restrict__ out) {
  extern __shared__ char smem[];
  const int wave = threadIdx.x >> 5, lane = threadIdx.x & 31;
  const int m = lane & 15, g = lane >> 4;
  char* wb = smem + (size_t)wave * 10240;
  half_t* h1s = (half_t*)wb;               // 16x128 f16 (4KB)
  half_t* h2s = (half_t*)(wb + 4096);      // 16x128 f16 (4KB)
  float*  tb  = (float*)(wb + 8192);       // 16x16 f32  (1KB)
  float*  ob  = (float*)(wb + 9216);       // 16x16 f32  (1KB)
  const int r0 = (blockIdx.x * 4 + wave) * 16;
  const int b  = r0 / NS;                  // uniform within wave (16 | 4096)
  const float* cvec = ctx + b * CTXD;

  if (lane < 16) {
    const float* zr = z + (size_t)(r0 + lane) * NP;
    for (int j = 0; j < NP; ++j) tb[lane * 16 + j] = zr[j];
    for (int j = NP; j < 16; ++j) tb[lane * 16 + j] = 0.f;
  }
  wfence();

  for (int stage = 0; stage < 2; ++stage) {
    const int split = (stage == 0) ? 5 : 6;
    const int rest  = (stage == 0) ? 6 : 5;
    const half_t* W1p = (stage == 0) ? W1pa : W1pb;
    const half_t* W2p = (stage == 0) ? W2pa : W2pb;
    const half_t* W3p = (stage == 0) ? W3pa : W3pb;
    const float* b1 = (stage == 0) ? b1a : b1b;
    const float* b2 = (stage == 0) ? b2a : b2b;
    const float* b3 = (stage == 0) ? b3a : b3b;

    // Build A = [theta1 | ctx | 0] as two K=32 fragments (K padded to 64).
    FragH a0, a1;
#pragma unroll
    for (int h = 0; h < 16; ++h) {
      int K0 = kmap(h, g), K1 = K0 + 32;
      float v0 = (K0 < split) ? tb[m * 16 + K0]
               : (K0 < split + CTXD) ? cvec[K0 - split] : 0.f;
      float v1 = (K1 < split + CTXD) ? cvec[K1 - split] : 0.f;
      a0.e[h] = (half_t)v0;
      a1.e[h] = (half_t)v1;
    }
    for (int n = 0; n < 8; ++n) {                         // h1 = relu(A@W1^T+b1)
      v8f acc = {};
      acc = wmma(a0.v, fragA_f16(W1p + (size_t)(n * 16) * 64, 64, lane), acc);
      acc = wmma(a1.v, fragA_f16(W1p + (size_t)(n * 16) * 64 + 32, 64, lane), acc);
      float bias = b1[n * 16 + m];
#pragma unroll
      for (int r = 0; r < 8; ++r) {
        float vv = acc[r] + bias;
        h1s[(r + 8 * g) * 128 + n * 16 + m] = (half_t)(vv > 0.f ? vv : 0.f);
      }
    }
    wfence();
    for (int n = 0; n < 8; ++n) {                         // h2 = relu(h1@W2^T+b2)
      v8f acc = {};
#pragma unroll
      for (int kk = 0; kk < 4; ++kk)
        acc = wmma(fragA_f16(h1s + kk * 32, 128, lane),
                   fragA_f16(W2p + (size_t)(n * 16) * 128 + kk * 32, 128, lane), acc);
      float bias = b2[n * 16 + m];
#pragma unroll
      for (int r = 0; r < 8; ++r) {
        float vv = acc[r] + bias;
        h2s[(r + 8 * g) * 128 + n * 16 + m] = (half_t)(vv > 0.f ? vv : 0.f);
      }
    }
    wfence();
    {                                                     // out = h2@W3^T+b3
      v8f acc = {};
#pragma unroll
      for (int kk = 0; kk < 4; ++kk)
        acc = wmma(fragA_f16(h2s + kk * 32, 128, lane),
                   fragA_f16(W3p + kk * 32, 128, lane), acc);
      float bias = (m < 2 * rest) ? b3[m] : 0.f;
#pragma unroll
      for (int r = 0; r < 8; ++r) ob[(r + 8 * g) * 16 + m] = acc[r] + bias;
    }
    wfence();
    if (lane < 16) {                                      // coupling update
      float* trow = tb + lane * 16;
      float* orow = ob + lane * 16;
      if (stage == 0) {   // theta = [theta2*exp(tanh(s))+t (6), theta1 (5)]
        float t1[5], nv[6];
        for (int j = 0; j < 5; ++j) t1[j] = trow[j];
        for (int j = 0; j < 6; ++j)
          nv[j] = trow[5 + j] * __expf(tanhf(orow[j])) + orow[6 + j];
        for (int j = 0; j < 6; ++j) trow[j] = nv[j];
        for (int j = 0; j < 5; ++j) trow[6 + j] = t1[j];
      } else {            // theta = [theta1 (6), theta2*exp(tanh(s))+t (5)]
        for (int j = 0; j < 5; ++j)
          trow[6 + j] = trow[6 + j] * __expf(tanhf(orow[j])) + orow[5 + j];
      }
    }
    wfence();
  }
  if (lane < 16) {
    const float* trow = tb + lane * 16;
    float* dst = out + (size_t)(r0 + lane) * NP;
    for (int j = 0; j < NP; ++j) dst[j] = trow[j];
  }
}

// ---------------------------------------------------------------------------
// Host orchestration. Input order assumes jax tree-flatten (sorted dict keys):
//  0 tokens, 1 mask, 2 z,
//  3 ctx_W, 4 ctx_b, 5 emb_W1, 6 emb_W2, 7 emb_b1, 8 emb_b2,
//  9..14 flow[0]{W1,W2,W3,b1,b2,b3}, 15..20 flow[1]{...},
//  21..32 layers[0]{Wf1,Wf2,Wo,Wqkv,bf1,bf2,bo,bqkv,ln1_b,ln1_g,ln2_b,ln2_g},
//  33..44 layers[1]{...}, 45 summary.
// ---------------------------------------------------------------------------
extern "C" void kernel_launch(void* const* d_in, const int* in_sizes, int n_in,
                              void* d_out, int out_size, void* d_ws, size_t ws_size,
                              hipStream_t stream) {
  (void)in_sizes; (void)n_in; (void)out_size; (void)ws_size;
  const float* tokens = (const float*)d_in[0];
  const unsigned char* mask = (const unsigned char*)d_in[1];  // jax bool (1B)
  const float* z      = (const float*)d_in[2];
  const float* ctx_W  = (const float*)d_in[3];
  const float* ctx_b  = (const float*)d_in[4];
  const float* emb_W1 = (const float*)d_in[5];
  const float* emb_W2 = (const float*)d_in[6];
  const float* emb_b1 = (const float*)d_in[7];
  const float* emb_b2 = (const float*)d_in[8];
  const float* fW1[2] = {(const float*)d_in[9],  (const float*)d_in[15]};
  const float* fW2[2] = {(const float*)d_in[10], (const float*)d_in[16]};
  const float* fW3[2] = {(const float*)d_in[11], (const float*)d_in[17]};
  const float* fb1[2] = {(const float*)d_in[12], (const float*)d_in[18]};
  const float* fb2[2] = {(const float*)d_in[13], (const float*)d_in[19]};
  const float* fb3[2] = {(const float*)d_in[14], (const float*)d_in[20]};
  struct Layer { const float *Wf1,*Wf2,*Wo,*Wqkv,*bf1,*bf2,*bo,*bqkv,*ln1b,*ln1g,*ln2b,*ln2g; };
  Layer lp[2];
  for (int l = 0; l < 2; ++l) {
    int base = 21 + 12 * l;
    lp[l].Wf1  = (const float*)d_in[base + 0];
    lp[l].Wf2  = (const float*)d_in[base + 1];
    lp[l].Wo   = (const float*)d_in[base + 2];
    lp[l].Wqkv = (const float*)d_in[base + 3];
    lp[l].bf1  = (const float*)d_in[base + 4];
    lp[l].bf2  = (const float*)d_in[base + 5];
    lp[l].bo   = (const float*)d_in[base + 6];
    lp[l].bqkv = (const float*)d_in[base + 7];
    lp[l].ln1b = (const float*)d_in[base + 8];
    lp[l].ln1g = (const float*)d_in[base + 9];
    lp[l].ln2b = (const float*)d_in[base + 10];
    lp[l].ln2g = (const float*)d_in[base + 11];
  }
  const float* summary = (const float*)d_in[45];
  float* out = (float*)d_out;

  // Workspace carve-up (~38 MB), all chunks 16B-aligned.
  char* w = (char*)d_ws;
  size_t off = 0;
  float*  xbuf  = (float*)(w + off);  off += (size_t)BATCH * SEQP * DMODEL * 4;
  half_t* xhbuf = (half_t*)(w + off); off += (size_t)BATCH * SEQP * DMODEL * 2;
  half_t* qbuf  = (half_t*)(w + off); off += (size_t)BATCH * SEQP * DMODEL * 2;
  half_t* kbuf  = (half_t*)(w + off); off += (size_t)BATCH * SEQP * DMODEL * 2;
  half_t* vTbuf = (half_t*)(w + off); off += (size_t)BATCH * DMODEL * SEQP * 2;
  half_t* obuf  = (half_t*)(w + off); off += (size_t)BATCH * SEQP * DMODEL * 2;
  float*  cbuf  = (float*)(w + off);  off += (size_t)BATCH * CTXD * 4;
  half_t* tok16 = (half_t*)(w + off); off += (size_t)BATCH * LTOK * 96 * 2;
  half_t* embW1p = (half_t*)(w + off); off += 256 * 96 * 2;
  half_t* embW2p = (half_t*)(w + off); off += 128 * 256 * 2;
  half_t* Wqkvp[2]; half_t* Wop[2]; half_t* Wf1p[2]; half_t* Wf2p[2];
  for (int l = 0; l < 2; ++l) {
    Wqkvp[l] = (half_t*)(w + off); off += 384 * 128 * 2;
    Wop[l]   = (half_t*)(w + off); off += 128 * 128 * 2;
    Wf1p[l]  = (half_t*)(w + off); off += 256 * 128 * 2;
    Wf2p[l]  = (half_t*)(w + off); off += 128 * 256 * 2;
  }
  half_t* W1p[2]; half_t* W2p[2]; half_t* W3p[2];
  for (int l = 0; l < 2; ++l) { W1p[l] = (half_t*)(w + off); off += 128 * 64 * 2; }
  for (int l = 0; l < 2; ++l) { W2p[l] = (half_t*)(w + off); off += 128 * 128 * 2; }
  for (int l = 0; l < 2; ++l) { W3p[l] = (half_t*)(w + off); off += 16 * 128 * 2; }

  hipMemsetAsync(d_ws, 0, off, stream);  // zero-fill incl. seq padding rows

  // One-time f32 -> f16 weight / token conversion.
  k_cvt<<<256, 256, 0, stream>>>(tokens, tok16, BATCH * LTOK * 96);
  k_cvt<<<64, 256, 0, stream>>>(emb_W1, embW1p, 256 * 96);
  k_cvt<<<64, 256, 0, stream>>>(emb_W2, embW2p, 128 * 256);
  for (int l = 0; l < 2; ++l) {
    k_cvt<<<64, 256, 0, stream>>>(lp[l].Wqkv, Wqkvp[l], 384 * 128);
    k_cvt<<<64, 256, 0, stream>>>(lp[l].Wo, Wop[l], 128 * 128);
    k_cvt<<<64, 256, 0, stream>>>(lp[l].Wf1, Wf1p[l], 256 * 128);
    k_cvt<<<64, 256, 0, stream>>>(lp[l].Wf2, Wf2p[l], 128 * 256);
  }
  k_pack<<<32, 256, 0, stream>>>(fW1[0], fW2[0], fW3[0], fW1[1], fW2[1], fW3[1],
                                 W1p[0], W2p[0], W3p[0], W1p[1], W2p[1], W3p[1]);

  k_embed<<<dim3(16, BATCH), 128, 16 * DFF * 2, stream>>>(
      tok16, embW1p, emb_b1, embW2p, emb_b2, summary, xbuf, xhbuf);

  for (int l = 0; l < 2; ++l) {
    k_qkv<<<dim3(17, BATCH), 128, 0, stream>>>(xhbuf, Wqkvp[l], lp[l].bqkv,
                                               qbuf, kbuf, vTbuf);
    k_attn<<<dim3(17, 2, BATCH), 32, 16 * SEQP * 4 + 16 * SEQP * 2, stream>>>(
        qbuf, kbuf, vTbuf, mask, obuf);
    k_projln<<<dim3(17, BATCH), 128, 16 * DMODEL * 4, stream>>>(
        obuf, Wop[l], lp[l].bo, lp[l].ln1g, lp[l].ln1b, xbuf, xhbuf);
    k_ffn<<<dim3(17, BATCH), 128, 16 * DFF * 2 + 16 * DMODEL * 4, stream>>>(
        Wf1p[l], lp[l].bf1, Wf2p[l], lp[l].bf2, lp[l].ln2g, lp[l].ln2b,
        xbuf, xhbuf);
  }

  k_ctx<<<BATCH, CTXD, 0, stream>>>(xbuf, ctx_W, ctx_b, cbuf);

  k_flow<<<NROWS / 64, 128, 4 * 10240, stream>>>(
      z, cbuf,
      W1p[0], W2p[0], W3p[0], fb1[0], fb2[0], fb3[0],
      W1p[1], W2p[1], W3p[1], fb1[1], fb2[1], fb3[1],
      out);
}
